// MultiHeadAttentionDistance_25099788877980
// MI455X (gfx1250) — matrix operand order repaired and checked by reading, not compiled, not run on hardware
//
#include <hip/hip_runtime.h>
#include <hip/hip_bf16.h>

// MultiHeadAttentionDistance for MI455X (gfx1250, wave32, WMMA).
// B=2, L=2048, D=1024, H=16, DH=64, sigma^2 = 64.
// Softmax identity: q_norm^2 cancels along the key axis, so
//   attn = softmax(q.k/64 - |k|^2/128)  (causal) = flash attention + key bias.
//
// Pipeline:
//   cvt (f32->f16 once) -> WMMA GEMMs with async global->LDS double-buffered
//   staging -> |k|^2 -> single-wave WMMA flash attention -> WMMA out-proj.

typedef _Float16 f16;
typedef __attribute__((ext_vector_type(16))) _Float16 v16h;
typedef __attribute__((ext_vector_type(8)))  _Float16 v8h;
typedef __attribute__((ext_vector_type(4)))  _Float16 v4h;
typedef __attribute__((ext_vector_type(8)))  float    v8f;

#define BN 2
#define LN 2048
#define DN 1024
#define HN 16
#define DH 64

static __device__ __forceinline__ v16h cat16(v8h lo, v8h hi) {
  return __builtin_shufflevector(lo, hi, 0,1,2,3,4,5,6,7,8,9,10,11,12,13,14,15);
}

// D = A x B + C, A/B f16, C/D f32, 16x16x32.
static __device__ __forceinline__ v8f wmma_f16(v16h a, v16h b, v8f c) {
  return __builtin_amdgcn_wmma_f32_16x16x32_f16(false, a, false, b, (short)0, c, false, false);
}

// Async 16-byte global -> LDS copy (ASYNCcnt-tracked, no VGPR round trip).
// lds_off: LDS byte address (low 32 bits of the flat shared pointer).
static __device__ __forceinline__ void async_copy_b128(unsigned lds_off,
                                                       const void* gptr) {
  asm volatile("global_load_async_to_lds_b128 %0, %1, off"
               :: "v"(lds_off), "v"((unsigned long long)(uintptr_t)gptr)
               : "memory");
}
static __device__ __forceinline__ void wait_async0() {
  asm volatile("s_wait_asynccnt 0x0" ::: "memory");
}

// ---------------------------------------------------------------------------
// Elementwise f32 -> f16 (n multiple of 4).
// ---------------------------------------------------------------------------
__global__ __launch_bounds__(256) void cvt_kernel(const float* __restrict__ src,
                                                  f16* __restrict__ dst, int n) {
  const int i = (blockIdx.x * 256 + threadIdx.x) * 4;
  if (i < n) {
    const float4 v = *(const float4*)(src + i);
    v4h h;
    h[0] = (f16)v.x; h[1] = (f16)v.y; h[2] = (f16)v.z; h[3] = (f16)v.w;
    *(v4h*)(dst + i) = h;
  }
}

// ---------------------------------------------------------------------------
// GEMM: out[m][n] = sum_k A[m][k] * W[n][k]   (M=4096, N=1024, K=1024)
// A, W already f16. Tiles staged via GLOBAL_LOAD_ASYNC_TO_LDS_B128 with
// ping-pong double buffering: next tile's copies fly during current WMMAs.
// MODE 0: store f16 row-major [4096][1024]           (Q, K projections)
// MODE 1: store f16 transposed per head [B,H,DH,L]   (V -> V^T for PV wmma)
// MODE 2: store f32 row-major [4096][1024]           (output projection)
// Block: 128 threads = 4 waves; 64x64 output tile; BK=32.
// ---------------------------------------------------------------------------
template <int MODE>
__global__ __launch_bounds__(128) void gemm_kernel(const f16* __restrict__ A,
                                                   const f16* __restrict__ W,
                                                   void* __restrict__ out) {
  __shared__ __align__(16) f16 At[2][64][32];
  __shared__ __align__(16) f16 Bt[2][64][32];

  const int mBase = blockIdx.y * 64;
  const int nBase = blockIdx.x * 64;
  const int t    = threadIdx.x;
  const int lane = t & 31;
  const int w    = t >> 5;        // wave id 0..3 -> M sub-tile
  const int lh   = lane >> 4;     // half-wave (0/1)
  const int ln   = lane & 15;

  // Staging: 64x32 halfs = 256 x 16B chunks per matrix; 2 chunks/thread/matrix.
  const int sr = t >> 2;          // chunk row 0..31 (and +32)
  const int sc = (t & 3) * 8;     // chunk col {0,8,16,24} halfs

  auto issue = [&](int buf, int kb) {
    const f16* a0 = A + (size_t)(mBase + sr) * DN + kb + sc;
    const f16* a1 = A + (size_t)(mBase + sr + 32) * DN + kb + sc;
    const f16* b0 = W + (size_t)(nBase + sr) * DN + kb + sc;
    const f16* b1 = W + (size_t)(nBase + sr + 32) * DN + kb + sc;
    async_copy_b128((unsigned)(uintptr_t)&At[buf][sr][sc],      a0);
    async_copy_b128((unsigned)(uintptr_t)&At[buf][sr + 32][sc], a1);
    async_copy_b128((unsigned)(uintptr_t)&Bt[buf][sr][sc],      b0);
    async_copy_b128((unsigned)(uintptr_t)&Bt[buf][sr + 32][sc], b1);
  };

  v8f acc[4] = {};                // 4 N-tiles of 16x16 f32

  issue(0, 0);
  for (int i = 0; i < DN / 32; ++i) {
    const int cur = i & 1;
    wait_async0();                // my copies for buf[cur] have landed
    __syncthreads();              // everyone's copies have landed
    if (i + 1 < DN / 32) issue(1 - cur, (i + 1) * 32);  // overlap next tile

    // A fragment 16x32: lane ln holds row (w*16+ln); halves map to
    // k = (h>>3)*16 + lh*8 + (h&7)  (ISA 7.12.2, 16-bit A 16x32)
    const int ar = w * 16 + ln;
    const v16h a = cat16(*(const v8h*)&At[cur][ar][lh * 8],
                         *(const v8h*)&At[cur][ar][16 + lh * 8]);
#pragma unroll
    for (int nt = 0; nt < 4; ++nt) {
      // B fragment 32x16: B[k][n] = W[nBase+nt*16+n][kb+k]; lane n = ln,
      // k = (h>>3)*16 + lh*8 + (h&7)  (same order as A).
      const int br = nt * 16 + ln;
      const int c0 = lh * 8;
      const v16h bf = cat16(*(const v8h*)&Bt[cur][br][c0],
                            *(const v8h*)&Bt[cur][br][c0 + 16]);
      acc[nt] = wmma_f16(a, bf, acc[nt]);
    }
    __syncthreads();              // done reading buf[cur] before it is refilled
  }

  // ---- epilogue; C/D layout: vgpr r -> row r + lh*8, lane ln -> col ----
#pragma unroll
  for (int nt = 0; nt < 4; ++nt) {
#pragma unroll
    for (int r = 0; r < 8; ++r) {
      const int gr = mBase + w * 16 + r + lh * 8;  // 0..4095  (= b*L + l)
      const int gc = nBase + nt * 16 + ln;         // 0..1023
      const float v = acc[nt][r];
      if (MODE == 0) {
        ((f16*)out)[(size_t)gr * DN + gc] = (f16)v;
      } else if (MODE == 1) {
        const int b = gr >> 11, l = gr & 2047, hh = gc >> 6, d = gc & 63;
        ((f16*)out)[(size_t)(((b * HN + hh) * DH + d)) * LN + l] = (f16)v;
      } else {
        ((float*)out)[(size_t)gr * DN + gc] = v;
      }
    }
  }
}

// ---------------------------------------------------------------------------
// |k|^2 per (b, h, l) from f16 K.
// ---------------------------------------------------------------------------
__global__ __launch_bounds__(256) void knorm_kernel(const f16* __restrict__ Kh,
                                                    float* __restrict__ kn2) {
  const int idx = blockIdx.x * 256 + threadIdx.x;  // = ((b*H + h)*L) + l
  const int l = idx & (LN - 1);
  const int hh = (idx >> 11) & (HN - 1);
  const int b = idx >> 15;
  const f16* p = Kh + (size_t)(b * LN + l) * DN + hh * DH;
  float s = 0.f;
#pragma unroll
  for (int d = 0; d < DH; ++d) {
    const float v = (float)p[d];
    s += v * v;
  }
  kn2[idx] = s;
}

// ---------------------------------------------------------------------------
// Flash attention, one wave per (16-row q tile, head, batch).
// Keys processed 32 at a time: S via 4 wmma, online softmax with half-wave
// shfl reductions, P re-laid out via LDS into an A fragment, then
// O += P x V via 4 wmma against V^T fragments.
// ---------------------------------------------------------------------------
__global__ __launch_bounds__(32) void attn_kernel(const f16* __restrict__ Qh,
                                                  const f16* __restrict__ Kh,
                                                  const f16* __restrict__ Vt,
                                                  const float* __restrict__ kn2,
                                                  f16* __restrict__ Oh) {
  const int qt = blockIdx.x, hh = blockIdx.y, b = blockIdx.z;
  const int lane = threadIdx.x & 31;
  const int lh = lane >> 4, ln = lane & 15;
  const int qBase = qt * 16;

  // Q fragments: two 16x32 A-fragments covering DH=64.
  const size_t rowQ = (size_t)(b * LN + qBase + ln) * DN + hh * DH;
  v16h qa[2];
#pragma unroll
  for (int c = 0; c < 2; ++c) {
    const f16* p = Qh + rowQ + c * 32 + lh * 8;
    qa[c] = cat16(*(const v8h*)p, *(const v8h*)(p + 16));
  }

  v8f o[4] = {};            // O accumulator: 16 x 64 (4 C-tiles)
  float m[8], lsum[8];
#pragma unroll
  for (int r = 0; r < 8; ++r) { m[r] = -1e30f; lsum[r] = 0.f; }

  __shared__ __align__(16) f16 Ph[16][32];

  const int nblk = (qBase + 16 + 31) >> 5;  // causal: keys < qBase+16
  for (int kb = 0; kb < nblk; ++kb) {
    const int keyBase = kb * 32;

    // ---- S = Q K^T : two 16x16 tiles (keys keyBase..+31) ----
    v8f s[2] = {};
#pragma unroll
    for (int tt = 0; tt < 2; ++tt) {
#pragma unroll
      for (int c = 0; c < 2; ++c) {
        // B[k][n] = K[keyBase+tt*16+n][c*32+k]; lane n = ln, k = (h>>3)*16 + lh*8 + (h&7)
        const f16* p = Kh + (size_t)(b * LN + keyBase + tt * 16 + ln) * DN +
                       hh * DH + c * 32 + lh * 8;
        const v16h bf = cat16(*(const v8h*)p, *(const v8h*)(p + 16));
        s[tt] = wmma_f16(qa[c], bf, s[tt]);
      }
    }

    // ---- scale 1/64, bias -|k|^2/128, causal mask ----
#pragma unroll
    for (int tt = 0; tt < 2; ++tt) {
      const int key = keyBase + tt * 16 + ln;
      const float bias = -0.0078125f * kn2[((b * HN + hh) * LN) + key];
      const int qrow0 = qBase + lh * 8;
#pragma unroll
      for (int r = 0; r < 8; ++r) {
        float v = s[tt][r] * 0.015625f + bias;
        if (key > qrow0 + r) v = -1e30f;
        s[tt][r] = v;
      }
    }

    // ---- online softmax ----
#pragma unroll
    for (int r = 0; r < 8; ++r) {
      float v = fmaxf(s[0][r], s[1][r]);
      v = fmaxf(v, __shfl_xor(v, 1, 32));
      v = fmaxf(v, __shfl_xor(v, 2, 32));
      v = fmaxf(v, __shfl_xor(v, 4, 32));
      v = fmaxf(v, __shfl_xor(v, 8, 32));
      const float mnew = fmaxf(m[r], v);
      const float sc = __expf(m[r] - mnew);
      m[r] = mnew;
      const float p0 = __expf(s[0][r] - mnew);
      const float p1 = __expf(s[1][r] - mnew);
      s[0][r] = p0; s[1][r] = p1;
      float rs = p0 + p1;
      rs += __shfl_xor(rs, 1, 32);
      rs += __shfl_xor(rs, 2, 32);
      rs += __shfl_xor(rs, 4, 32);
      rs += __shfl_xor(rs, 8, 32);
      lsum[r] = lsum[r] * sc + rs;
#pragma unroll
      for (int nt = 0; nt < 4; ++nt) o[nt][r] *= sc;
    }

    // ---- P: C-layout f32 -> A-layout f16 via LDS ----
#pragma unroll
    for (int tt = 0; tt < 2; ++tt)
#pragma unroll
      for (int r = 0; r < 8; ++r)
        Ph[r + lh * 8][tt * 16 + ln] = (f16)s[tt][r];
    asm volatile("s_wait_dscnt 0" ::: "memory");
    const v16h pa = cat16(*(const v8h*)&Ph[ln][lh * 8],
                          *(const v8h*)&Ph[ln][16 + lh * 8]);

    // ---- O += P x V : 4 wmma against V^T [B,H,DH,L] ----
#pragma unroll
    for (int nt = 0; nt < 4; ++nt) {
      // B[k][n] = V[keyBase+k][nt*16+n]; lane n = ln, k = (h>>3)*16 + lh*8 + (h&7)
      const f16* p = Vt + (size_t)((b * HN + hh) * DH + nt * 16 + ln) * LN +
                     keyBase + lh * 8;
      const v16h bf = cat16(*(const v8h*)p, *(const v8h*)(p + 16));
      o[nt] = wmma_f16(pa, bf, o[nt]);
    }
  }

  // ---- normalize and store Oh (f16, [B,L,D] with head offset) ----
#pragma unroll
  for (int r = 0; r < 8; ++r) {
    const float inv = 1.0f / lsum[r];
    const int gl = qBase + lh * 8 + r;
    const size_t base = (size_t)(b * LN + gl) * DN + hh * DH + ln;
#pragma unroll
    for (int nt = 0; nt < 4; ++nt)
      Oh[base + nt * 16] = (f16)(o[nt][r] * inv);
  }
}

// ---------------------------------------------------------------------------
extern "C" void kernel_launch(void* const* d_in, const int* in_sizes, int n_in,
                              void* d_out, int out_size, void* d_ws, size_t ws_size,
                              hipStream_t stream) {
  const float* x  = (const float*)d_in[0];
  // d_in[1] = mask: exactly tril -> implemented analytically, not read.
  const float* Wq = (const float*)d_in[2];
  const float* Wk = (const float*)d_in[3];
  const float* Wv = (const float*)d_in[4];
  const float* Wo = (const float*)d_in[5];

  char* ws = (char*)d_ws;
  f16*   Xh  = (f16*)(ws);                    //  8 MB [B*L, D] f16
  f16*   Wqh = (f16*)(ws + (8u  << 20));      //  2 MB
  f16*   Wkh = (f16*)(ws + (10u << 20));      //  2 MB
  f16*   Wvh = (f16*)(ws + (12u << 20));      //  2 MB
  f16*   Woh = (f16*)(ws + (14u << 20));      //  2 MB
  f16*   Qh  = (f16*)(ws + (16u << 20));      //  8 MB [B*L, D] f16
  f16*   Kh  = (f16*)(ws + (24u << 20));      //  8 MB [B*L, D] f16
  f16*   Vt  = (f16*)(ws + (32u << 20));      //  8 MB [B,H,DH,L] f16
  f16*   Oh  = (f16*)(ws + (40u << 20));      //  8 MB [B*L, D] f16
  float* kn2 = (float*)(ws + (48u << 20));    // 256 KB [B,H,L] f32

  const int nx = BN * LN * DN;                // 4M
  const int nw = DN * DN;                     // 1M
  cvt_kernel<<<nx / 4 / 256, 256, 0, stream>>>(x,  Xh,  nx);
  cvt_kernel<<<nw / 4 / 256, 256, 0, stream>>>(Wq, Wqh, nw);
  cvt_kernel<<<nw / 4 / 256, 256, 0, stream>>>(Wk, Wkh, nw);
  cvt_kernel<<<nw / 4 / 256, 256, 0, stream>>>(Wv, Wvh, nw);
  cvt_kernel<<<nw / 4 / 256, 256, 0, stream>>>(Wo, Woh, nw);

  const dim3 gg(DN / 64, (BN * LN) / 64);     // (16, 64)
  const dim3 gb(128);
  gemm_kernel<0><<<gg, gb, 0, stream>>>(Xh, Wqh, Qh);
  gemm_kernel<0><<<gg, gb, 0, stream>>>(Xh, Wkh, Kh);
  gemm_kernel<1><<<gg, gb, 0, stream>>>(Xh, Wvh, Vt);
  knorm_kernel<<<(BN * HN * LN) / 256, 256, 0, stream>>>(Kh, kn2);
  attn_kernel<<<dim3(LN / 16, HN, BN), 32, 0, stream>>>(Qh, Kh, Vt, kn2, Oh);
  gemm_kernel<2><<<gg, gb, 0, stream>>>(Oh, Woh, d_out);
}
